// BiLSTM_CRF_88192858456349
// MI455X (gfx1250) — matrix operand order repaired and checked by reading, not compiled, not run on hardware
//
#include <hip/hip_runtime.h>
#include <hip/hip_bf16.h>

#define Lseq   8192
#define Edim   300
#define EPAD   320
#define Hdim   256
#define G4H    1024
#define NCOL   2048   // forward 1024 gates | backward 1024 gates
#define NTAG   5

typedef __attribute__((ext_vector_type(16))) __bf16    v16bf;
typedef __attribute__((ext_vector_type(8)))  float     v8f;
typedef __attribute__((ext_vector_type(8)))  unsigned  v8u;

static __device__ __forceinline__ unsigned short f2bf(float f) {
    unsigned u = __builtin_bit_cast(unsigned, f);
    u += 0x7FFFu + ((u >> 16) & 1u);          // round-to-nearest-even
    return (unsigned short)(u >> 16);
}
static __device__ __forceinline__ float bf2f(unsigned short h) {
    unsigned u = ((unsigned)h) << 16;
    return __builtin_bit_cast(float, u);
}
static __device__ __forceinline__ float sigmf(float x) {
    return 1.0f / (1.0f + __expf(-x));
}

// ---------------- prep: W_ih (both dirs) -> bf16 [NCOL][EPAD], zero-padded K ----
__global__ void prep_wih(const float* __restrict__ Wf, const float* __restrict__ Wb,
                         unsigned short* __restrict__ Wbf) {
    int idx = blockIdx.x * blockDim.x + threadIdx.x;
    if (idx >= NCOL * EPAD) return;
    int n = idx / EPAD, k = idx % EPAD;
    float v = 0.0f;
    if (k < Edim) v = (n < G4H) ? Wf[n * Edim + k] : Wb[(n - G4H) * Edim + k];
    Wbf[idx] = f2bf(v);
}

// ---------------- prep: W_hh (both dirs) -> bf16 [2][G4H][Hdim] ------------------
__global__ void prep_whh(const float* __restrict__ Wf, const float* __restrict__ Wb,
                         unsigned short* __restrict__ Wbf) {
    int idx = blockIdx.x * blockDim.x + threadIdx.x;
    if (idx >= 2 * G4H * Hdim) return;
    float v = (idx < G4H * Hdim) ? Wf[idx] : Wb[idx - G4H * Hdim];
    Wbf[idx] = f2bf(v);
}

// ---------------- embedding gather -> bf16 [Lseq][EPAD] --------------------------
__global__ void gather_embed(const int* __restrict__ sent, const float* __restrict__ emb,
                             unsigned short* __restrict__ Xbf) {
    int idx = blockIdx.x * blockDim.x + threadIdx.x;
    if (idx >= Lseq * EPAD) return;
    int l = idx / EPAD, k = idx % EPAD;
    float v = (k < Edim) ? emb[(size_t)sent[l] * Edim + k] : 0.0f;
    Xbf[idx] = f2bf(v);
}

// ---------------- WMMA GEMM: P[Lseq][NCOL] = X[L,EPAD] @ W^T + bias --------------
// one wave computes a 16x64 output strip (4 N-tiles): A fragment reused 4x per K-step
__global__ void xproj_wmma(const unsigned short* __restrict__ Xbf,
                           const unsigned short* __restrict__ Wbf,
                           const float* __restrict__ bf_, const float* __restrict__ bb_,
                           float* __restrict__ P) {
    int wave  = (blockIdx.x * blockDim.x + threadIdx.x) >> 5;
    int lane  = threadIdx.x & 31;
    int mtile = wave >> 5;          // 32 n-groups (of 64 cols) per row-tile
    int ngrp  = wave & 31;
    int khalf = lane >> 4;
    int m     = lane & 15;
    int arow  = mtile * 16 + m;
    int ncol0 = ngrp * 64 + (lane & 15);

    const unsigned short* ap  = Xbf + (size_t)arow  * EPAD;
    const unsigned short* bp0 = Wbf + (size_t)ncol0 * EPAD;

    v8f c[4];
#pragma unroll
    for (int s = 0; s < 4; ++s) {
        int nc = ncol0 + 16 * s;
        float bias = (nc < G4H) ? bf_[nc] : bb_[nc - G4H];
#pragma unroll
        for (int r = 0; r < 8; ++r) c[s][r] = bias;
    }

    for (int kb = 0; kb < EPAD; kb += 32) {
        v8u au;
#pragma unroll
        for (int jj = 0; jj < 8; ++jj) {
            // A 16x32 layout: element pair (2jj,2jj+1) -> k = kb + 16*(jj>>2) + 8*khalf + 2*(jj&3)
            int ka = kb + 16 * (jj >> 2) + 8 * khalf + 2 * (jj & 3);
            au[jj] = *(const unsigned*)(ap + ka);
        }
        v16bf a = __builtin_bit_cast(v16bf, au);
#pragma unroll
        for (int s = 0; s < 4; ++s) {
            const unsigned short* bp = bp0 + (size_t)s * 16 * EPAD;
            v8u bu;
#pragma unroll
            for (int jj = 0; jj < 8; ++jj) {
                // B 32x16 layout: element pair -> k = kb + 16*(jj>>2) + 8*khalf + 2*(jj&3)
                int kc = kb + 16 * (jj >> 2) + 8 * khalf + 2 * (jj & 3);
                bu[jj] = *(const unsigned*)(bp + kc);
            }
            v16bf b = __builtin_bit_cast(v16bf, bu);
            c[s] = __builtin_amdgcn_wmma_f32_16x16x32_bf16(false, a, false, b,
                                                           (short)0, c[s], false, false);
        }
    }
#pragma unroll
    for (int s = 0; s < 4; ++s) {
#pragma unroll
        for (int r = 0; r < 8; ++r) {
            int row = mtile * 16 + 8 * khalf + r;   // C/D layout: M = r + 8*(lane>>4)
            P[(size_t)row * NCOL + ncol0 + 16 * s] = c[s][r];
        }
    }
}

// ---------------- sequential LSTM scan, one block (32 waves) per direction -------
__global__ void __launch_bounds__(1024, 1)
lstm_scan(const float* __restrict__ P, const unsigned short* __restrict__ Whh,
          const float* __restrict__ h0, const float* __restrict__ c0,
          float* __restrict__ hcat) {
    int dir = blockIdx.x;     // 0 = forward, 1 = backward
    int tid = threadIdx.x;    // 0..1023, owns gate row `tid`
    __shared__ __align__(16) float h_sh[Hdim];
    __shared__ float g_sh[G4H];

    const unsigned short* wrow = Whh + ((size_t)(dir * G4H + tid)) * Hdim;
    float c = 0.0f;
    if (tid < Hdim) {
        h_sh[tid] = h0[dir * Hdim + tid];
        c         = c0[dir * Hdim + tid];
    }
    __syncthreads();

    const float4* h4 = (const float4*)h_sh;
    for (int step = 0; step < Lseq; ++step) {
        int l = (dir == 0) ? step : (Lseq - 1 - step);
        const float* prow = P + (size_t)l * NCOL + dir * G4H;
        float acc = prow[tid];
        if (step + 1 < Lseq) {
            int ln = (dir == 0) ? (l + 1) : (l - 1);
            __builtin_prefetch(P + (size_t)ln * NCOL + dir * G4H + tid, 0, 1);
        }
#pragma unroll 8
        for (int k4 = 0; k4 < Hdim / 4; ++k4) {
            uint2  u = *(const uint2*)(wrow + 4 * k4);    // 4 bf16 weights
            float4 h = h4[k4];                            // ds_load_b128 broadcast
            acc += bf2f((unsigned short)(u.x & 0xFFFFu)) * h.x
                 + bf2f((unsigned short)(u.x >> 16))     * h.y
                 + bf2f((unsigned short)(u.y & 0xFFFFu)) * h.z
                 + bf2f((unsigned short)(u.y >> 16))     * h.w;
        }
        g_sh[tid] = acc;
        __syncthreads();
        if (tid < Hdim) {
            float ig = sigmf(g_sh[tid]);
            float fg = sigmf(g_sh[Hdim + tid]);
            float gg = tanhf(g_sh[2 * Hdim + tid]);
            float og = sigmf(g_sh[3 * Hdim + tid]);
            c = fg * c + ig * gg;
            float h = og * tanhf(c);
            h_sh[tid] = h;
            hcat[(size_t)l * (2 * Hdim) + dir * Hdim + tid] = h;
        }
        __syncthreads();
    }
}

// ---------------- emission[l][t] = hcat[l] . W_out[t] + b_out[t] -----------------
__global__ void emission_kernel(const float* __restrict__ hcat, const float* __restrict__ Wo,
                                const float* __restrict__ bo, float* __restrict__ em) {
    int l    = blockIdx.x;
    int tag  = threadIdx.x >> 5;   // blockDim = 160 -> 5 waves, one per tag
    int lane = threadIdx.x & 31;
    const float* h = hcat + (size_t)l * (2 * Hdim);
    const float* w = Wo + (size_t)tag * (2 * Hdim);
    float s = 0.0f;
    for (int k = lane; k < 2 * Hdim; k += 32) s += h[k] * w[k];
#pragma unroll
    for (int off = 16; off > 0; off >>= 1) s += __shfl_down(s, off, 32);
    if (lane == 0) em[l * NTAG + tag] = s + bo[tag];
}

// ---------------- Viterbi decode: single wave, backpointers in LDS ---------------
__global__ void viterbi_kernel(const float* __restrict__ em, const float* __restrict__ start,
                               const float* __restrict__ endv, const float* __restrict__ trans,
                               float* __restrict__ out) {
    __shared__ unsigned char bps[Lseq * NTAG];
    int lane = threadIdx.x;
    int t = (lane < NTAG) ? lane : 0;
    float tr[NTAG];
#pragma unroll
    for (int p = 0; p < NTAG; ++p) tr[p] = trans[p * NTAG + t];

    float score = start[t] + em[t];
    for (int s = 1; s < Lseq; ++s) {
        float e = em[s * NTAG + t];
        float best = -3.4e38f;
        int bp = 0;
#pragma unroll
        for (int p = 0; p < NTAG; ++p) {
            float v = __shfl(score, p, 32) + tr[p];
            if (v > best) { best = v; bp = p; }
        }
        if (lane < NTAG) bps[(s - 1) * NTAG + t] = (unsigned char)bp;
        score = best + e;
    }
    score += endv[t];
    float s0 = __shfl(score, 0, 32), s1 = __shfl(score, 1, 32), s2 = __shfl(score, 2, 32);
    float s3 = __shfl(score, 3, 32), s4 = __shfl(score, 4, 32);
    __syncthreads();
    if (lane == 0) {
        int last = 0; float bv = s0;
        if (s1 > bv) { bv = s1; last = 1; }
        if (s2 > bv) { bv = s2; last = 2; }
        if (s3 > bv) { bv = s3; last = 3; }
        if (s4 > bv) { bv = s4; last = 4; }
        out[Lseq - 1] = (float)last;
        int cur = last;
        for (int s = Lseq - 2; s >= 0; --s) {
            cur = bps[s * NTAG + cur];
            out[s] = (float)cur;
        }
    }
}

extern "C" void kernel_launch(void* const* d_in, const int* in_sizes, int n_in,
                              void* d_out, int out_size, void* d_ws, size_t ws_size,
                              hipStream_t stream) {
    const int*   sent  = (const int*)  d_in[0];
    const float* embed = (const float*)d_in[1];
    const float* Wih_f = (const float*)d_in[2];
    const float* Whh_f = (const float*)d_in[3];
    const float* b_f   = (const float*)d_in[4];
    const float* Wih_b = (const float*)d_in[5];
    const float* Whh_b = (const float*)d_in[6];
    const float* b_b   = (const float*)d_in[7];
    const float* h0    = (const float*)d_in[8];
    const float* c0    = (const float*)d_in[9];
    const float* W_out = (const float*)d_in[10];
    const float* b_out = (const float*)d_in[11];
    const float* start = (const float*)d_in[12];
    const float* endv  = (const float*)d_in[13];
    const float* trans = (const float*)d_in[14];
    float* out = (float*)d_out;

    char* w = (char*)d_ws;
    unsigned short* Xbf    = (unsigned short*)(w);                       // 5,242,880 B
    unsigned short* Wih_bf = (unsigned short*)(w + 5242880);             // 1,310,720 B
    unsigned short* Whh_bf = (unsigned short*)(w + 6553600);             // 1,048,576 B
    float*          P      = (float*)         (w + 7602176);             // 67,108,864 B
    float*          hcat   = (float*)         (w + 74711040);            // 16,777,216 B
    float*          em     = (float*)         (w + 91488256);            // 163,840 B

    prep_wih   <<<(NCOL * EPAD + 255) / 256, 256, 0, stream>>>(Wih_f, Wih_b, Wih_bf);
    prep_whh   <<<(2 * G4H * Hdim + 255) / 256, 256, 0, stream>>>(Whh_f, Whh_b, Whh_bf);
    gather_embed<<<(Lseq * EPAD + 255) / 256, 256, 0, stream>>>(sent, embed, Xbf);

    // 512 row-tiles * 32 n-groups = 16384 waves; 8 waves / 256-thread block
    xproj_wmma <<<16384 / 8, 256, 0, stream>>>(Xbf, Wih_bf, b_f, b_b, P);

    lstm_scan  <<<2, 1024, 0, stream>>>(P, Whh_bf, h0, c0, hcat);
    emission_kernel<<<Lseq, 32 * NTAG, 0, stream>>>(hcat, W_out, b_out, em);
    viterbi_kernel <<<1, 32, 0, stream>>>(em, start, endv, trans, out);
}
